// PointTransformerLayer_70153995813099
// MI455X (gfx1250) — matrix operand order repaired and checked by reading, not compiled, not run on hardware
//
#include <hip/hip_runtime.h>
#include <hip/hip_bf16.h>

typedef __attribute__((ext_vector_type(16))) _Float16 v16h;
typedef __attribute__((ext_vector_type(8)))  _Float16 v8h;
typedef __attribute__((ext_vector_type(8)))  float    v8f;

union V16 { v16h v; v8h h[2]; };

#define DIMC 128
#define KNN  16

// ---------------------------------------------------------------------------
// f32 -> f16 feature conversion
// ---------------------------------------------------------------------------
__global__ __launch_bounds__(256) void cvt_feats_kernel(const float* __restrict__ f,
                                                        _Float16* __restrict__ o, int n) {
    int i = blockIdx.x * 256 + threadIdx.x;
    if (i < n) o[i] = (_Float16)f[i];
}

// Convert 6 weight matrices [128x128] f32 row-major (W[k][n]) into f16
// transposed (WT[n][k]) so WMMA B-fragments are contiguous 128-bit loads.
// Order: wq, wk, wv, pos_w2, attn_w1, attn_w2.
__global__ __launch_bounds__(256) void cvt_weights_kernel(
    const float* __restrict__ w0, const float* __restrict__ w1,
    const float* __restrict__ w2, const float* __restrict__ w3,
    const float* __restrict__ w4, const float* __restrict__ w5,
    _Float16* __restrict__ o) {
    int i = blockIdx.x * 256 + threadIdx.x;   // 6*16384 total
    int mat = i >> 14;
    int e   = i & 16383;
    int k   = e >> 7;
    int n   = e & 127;
    const float* src = (mat == 0) ? w0 : (mat == 1) ? w1 : (mat == 2) ? w2
                     : (mat == 3) ? w3 : (mat == 4) ? w4 : w5;
    o[(size_t)mat * 16384 + (size_t)n * 128 + k] = (_Float16)src[(size_t)k * 128 + n];
}

// ---------------------------------------------------------------------------
// KNN: one thread per query point, candidates tiled through LDS,
// per-thread top-16 register insertion (ascending distance).
// ---------------------------------------------------------------------------
__global__ __launch_bounds__(256) void knn_kernel(const float* __restrict__ xyz,
                                                  int* __restrict__ idxOut, int Npts) {
    __shared__ float sx[256 * 3];
    int q    = blockIdx.x * 256 + threadIdx.x;      // global point id
    int b    = q / Npts;
    int base = b * Npts;

    float qx = xyz[(size_t)q * 3 + 0];
    float qy = xyz[(size_t)q * 3 + 1];
    float qz = xyz[(size_t)q * 3 + 2];

    float dist[KNN];
    int   ind[KNN];
#pragma unroll
    for (int t = 0; t < KNN; ++t) { dist[t] = 3.0e38f; ind[t] = 0; }

    for (int tile = 0; tile < Npts; tile += 256) {
        __syncthreads();
        int src = base + tile + threadIdx.x;
        sx[threadIdx.x * 3 + 0] = xyz[(size_t)src * 3 + 0];
        sx[threadIdx.x * 3 + 1] = xyz[(size_t)src * 3 + 1];
        sx[threadIdx.x * 3 + 2] = xyz[(size_t)src * 3 + 2];
        __syncthreads();
        for (int j = 0; j < 256; ++j) {
            float dx = qx - sx[j * 3 + 0];
            float dy = qy - sx[j * 3 + 1];
            float dz = qz - sx[j * 3 + 2];
            float d2 = dx * dx + dy * dy + dz * dz;
            if (d2 < dist[KNN - 1]) {           // guarded insertion (rare path)
                float d = d2; int id = tile + j;
#pragma unroll
                for (int t = 0; t < KNN; ++t) {
                    if (d < dist[t]) {
                        float td = dist[t]; int ti = ind[t];
                        dist[t] = d; ind[t] = id; d = td; id = ti;
                    }
                }
            }
        }
    }
#pragma unroll
    for (int t = 0; t < KNN; ++t) idxOut[(size_t)q * KNN + t] = ind[t];
}

// ---------------------------------------------------------------------------
// QKV projections: [R x 128] @ [128 x 128] x3 with v_wmma_f32_16x16x32_f16.
// Block = 8 waves; block owns one 16-row tile; each wave owns ONE 16-column
// tile and loops q/k/v fully unrolled (compile-time weight/bias/out pointers,
// no cndmask address chains). Per matrix: one 8-load clause, then 4 WMMAs on
// 4 *independent* accumulators (zero RAW hazards, zero v_nops), tree-added.
// ---------------------------------------------------------------------------
__global__ __launch_bounds__(256) void qkv_kernel(
    const _Float16* __restrict__ feats,   // [R][128] f16
    const _Float16* __restrict__ wT,      // [3][128][128] f16 transposed (N-major)
    const float* __restrict__ bq, const float* __restrict__ bk, const float* __restrict__ bv,
    float* __restrict__ qb, float* __restrict__ kb, float* __restrict__ vb) {
    int wave = threadIdx.x >> 5;
    int lane = threadIdx.x & 31;
    int nl   = lane & 15;
    int hi   = lane >> 4;
    int m0   = blockIdx.x * 16;
    int m    = m0 + nl;                   // A row for this lane
    int n0   = wave * 16;                 // this wave's output column tile

    // Warm the L2 with the whole f16 weight arena (6 * 32KB = 1536 lines).
    {
        int gid = blockIdx.x * 256 + threadIdx.x;
        if (gid < 6 * DIMC * DIMC * 2 / 128)
            __builtin_prefetch(((const char*)wT) + (size_t)gid * 128, 0, 3);
    }

    // A fragments for the 4 K-chunks (layout: ISA 7.12.2, 16-bit A 16x32)
    V16 a[4];
#pragma unroll
    for (int kc4 = 0; kc4 < 4; ++kc4) {
        int kc = kc4 * 32;
        a[kc4].h[0] = *(const v8h*)(feats + (size_t)m * 128 + kc + hi * 8);
        a[kc4].h[1] = *(const v8h*)(feats + (size_t)m * 128 + kc + 16 + hi * 8);
    }

#pragma unroll
    for (int mat = 0; mat < 3; ++mat) {   // unrolled: pointers are compile-time
        const _Float16* wt     = wT + (size_t)mat * 16384;
        const float*    bias_p = (mat == 0) ? bq : (mat == 1) ? bk : bv;
        float*          out_p  = (mat == 0) ? qb : (mat == 1) ? kb : vb;

        // All 4 B fragments in one load clause.
        V16 bf[4];
#pragma unroll
        for (int kc4 = 0; kc4 < 4; ++kc4) {
            int kc = kc4 * 32;
            bf[kc4].h[0] = *(const v8h*)(wt + (size_t)(n0 + nl) * 128 + kc + hi * 8);
            bf[kc4].h[1] = *(const v8h*)(wt + (size_t)(n0 + nl) * 128 + kc + 16 + hi * 8);
        }
        // 4 fully independent WMMAs -> no WMMA->WMMA RAW, no v_nops.
        v8f p0 = {}, p1 = {}, p2 = {}, p3 = {};
        p0 = __builtin_amdgcn_wmma_f32_16x16x32_f16(false, a[0].v, false, bf[0].v, (short)0, p0, false, false);
        p1 = __builtin_amdgcn_wmma_f32_16x16x32_f16(false, a[1].v, false, bf[1].v, (short)0, p1, false, false);
        p2 = __builtin_amdgcn_wmma_f32_16x16x32_f16(false, a[2].v, false, bf[2].v, (short)0, p2, false, false);
        p3 = __builtin_amdgcn_wmma_f32_16x16x32_f16(false, a[3].v, false, bf[3].v, (short)0, p3, false, false);
        v8f acc = (p0 + p1) + (p2 + p3);

        float bias = bias_p[n0 + nl];
#pragma unroll
        for (int j = 0; j < 8; ++j) {
            int row = m0 + j + hi * 8;    // C layout: vgpr j -> row M=j (+8 for hi half)
            out_p[(size_t)row * 128 + n0 + nl] = acc[j] + bias;
        }
    }
}

// ---------------------------------------------------------------------------
// One 16x128 activation tile (LDS, f16) times a 128x128 transposed weight
// matrix (global, f16) -> 8 f32 accumulators. K-chunk-outer / tile-inner so
// each chunk issues 8 *independent* back-to-back WMMAs after one load clause.
// ---------------------------------------------------------------------------
__device__ __forceinline__ void tile_gemm_128(const _Float16* __restrict__ sSrc,
                                              const _Float16* __restrict__ wt,
                                              int nl, int hi, v8f acc[8]) {
    V16 af[4];
#pragma unroll
    for (int kc4 = 0; kc4 < 4; ++kc4) {
        int kc = kc4 * 32;
        af[kc4].h[0] = *(const v8h*)(sSrc + nl * 128 + kc + hi * 8);
        af[kc4].h[1] = *(const v8h*)(sSrc + nl * 128 + kc + 16 + hi * 8);
    }
#pragma unroll
    for (int kc4 = 0; kc4 < 4; ++kc4) {
        int kc = kc4 * 32;
        V16 bf[8];
#pragma unroll
        for (int t = 0; t < 8; ++t) {
            bf[t].h[0] = *(const v8h*)(wt + (size_t)(t * 16 + nl) * 128 + kc + hi * 8);
            bf[t].h[1] = *(const v8h*)(wt + (size_t)(t * 16 + nl) * 128 + kc + 16 + hi * 8);
        }
#pragma unroll
        for (int t = 0; t < 8; ++t)
            acc[t] = __builtin_amdgcn_wmma_f32_16x16x32_f16(
                false, af[kc4].v, false, bf[t].v, (short)0, acc[t], false, false);
    }
}

// ---------------------------------------------------------------------------
// Fused per-point attention: one wave32 per point.
// gather -> pos-MLP L1 (VALU) -> delta GEMM (WMMA) -> attn L1 (WMMA+relu)
// -> attn L2 (WMMA) -> per-column softmax over 16 neighbors -> weighted sum.
// ---------------------------------------------------------------------------
__global__ __launch_bounds__(32) void pt_attn_kernel(
    const float* __restrict__ xyz, const int* __restrict__ idx,
    const float* __restrict__ qb, const float* __restrict__ kb, const float* __restrict__ vb,
    const _Float16* __restrict__ posw2T, const _Float16* __restrict__ aw1T,
    const _Float16* __restrict__ aw2T,
    const float* __restrict__ posw1, const float* __restrict__ posb1,
    const float* __restrict__ posb2, const float* __restrict__ ab1,
    const float* __restrict__ ab2,
    float* __restrict__ out, int Npts) {
    __shared__ float    sA[16 * 128];   // q - k_g, later attn logits (f32)
    __shared__ float    sB[16 * 128];   // v_g, later v_g + delta (f32)
    __shared__ _Float16 sH[16 * 128];   // f16 activation tile (WMMA A operand)
    __shared__ float    sPD[16 * 4];    // pos_diff, padded rows
    __shared__ int      sIdx[16];

    int lane = threadIdx.x;
    int r    = blockIdx.x;
    int b    = r / Npts;
    int base = b * Npts;
    int nl   = lane & 15;
    int hi   = lane >> 4;

    if (lane < 16) {
        int nb = idx[(size_t)r * KNN + lane];
        sIdx[lane] = nb;
        const float* g = xyz + (size_t)(base + nb) * 3;
        sPD[lane * 4 + 0] = xyz[(size_t)r * 3 + 0] - g[0];
        sPD[lane * 4 + 1] = xyz[(size_t)r * 3 + 1] - g[1];
        sPD[lane * 4 + 2] = xyz[(size_t)r * 3 + 2] - g[2];
    }
    __syncthreads();

    // Per-lane constants: each lane owns channels {lane, lane+32, lane+64, lane+96}.
    float qv[4], pb1v[4], pw1v[3][4];
#pragma unroll
    for (int cc = 0; cc < 4; ++cc) {
        int c = lane + cc * 32;
        qv[cc]      = qb[(size_t)r * 128 + c];
        pb1v[cc]    = posb1[c];
        pw1v[0][cc] = posw1[c];
        pw1v[1][cc] = posw1[128 + c];
        pw1v[2][cc] = posw1[256 + c];
    }

    // Gather k_g/v_g and run pos-MLP layer 1 (3->128, VALU).
    for (int kk = 0; kk < KNN; ++kk) {
        int gi = base + sIdx[kk];
        float pdx = sPD[kk * 4 + 0];
        float pdy = sPD[kk * 4 + 1];
        float pdz = sPD[kk * 4 + 2];
#pragma unroll
        for (int cc = 0; cc < 4; ++cc) {
            int c = lane + cc * 32;
            int e = kk * 128 + c;
            sA[e] = qv[cc] - kb[(size_t)gi * 128 + c];
            sB[e] = vb[(size_t)gi * 128 + c];
            float acc = pb1v[cc] + pdx * pw1v[0][cc] + pdy * pw1v[1][cc] + pdz * pw1v[2][cc];
            sH[e] = (_Float16)fmaxf(acc, 0.0f);
        }
    }
    __syncthreads();

    v8f acc[8];

    // ---- delta = relu(h1) @ pos_w2 (WMMA); bias added at writeback ----
#pragma unroll
    for (int t = 0; t < 8; ++t) acc[t] = (v8f){};
    tile_gemm_128(sH, posw2T, nl, hi, acc);
    __syncthreads();
    // writeback: attn_in = (q - k_g) + delta  ->  sH (f16);  v_g + delta -> sB
#pragma unroll
    for (int t = 0; t < 8; ++t) {
        int col  = t * 16 + nl;
        float bias = posb2[col];
#pragma unroll
        for (int j = 0; j < 8; ++j) {
            int row = j + hi * 8;
            int e   = row * 128 + col;
            float d = acc[t][j] + bias;
            sH[e] = (_Float16)(sA[e] + d);
            sB[e] = sB[e] + d;
        }
    }
    __syncthreads();

    // ---- attn layer 1: relu(attn_in @ attn_w1 + b1) -> sH (f16) ----
#pragma unroll
    for (int t = 0; t < 8; ++t) acc[t] = (v8f){};
    tile_gemm_128(sH, aw1T, nl, hi, acc);
    __syncthreads();
#pragma unroll
    for (int t = 0; t < 8; ++t) {
        int col = t * 16 + nl;
        float bias = ab1[col];
#pragma unroll
        for (int j = 0; j < 8; ++j) {
            int row = j + hi * 8;
            sH[row * 128 + col] = (_Float16)fmaxf(acc[t][j] + bias, 0.0f);
        }
    }
    __syncthreads();

    // ---- attn layer 2: logits -> sA (f32) ----
#pragma unroll
    for (int t = 0; t < 8; ++t) acc[t] = (v8f){};
    tile_gemm_128(sH, aw2T, nl, hi, acc);
    __syncthreads();
#pragma unroll
    for (int t = 0; t < 8; ++t) {
        int col = t * 16 + nl;
        float bias = ab2[col];
#pragma unroll
        for (int j = 0; j < 8; ++j) {
            int row = j + hi * 8;
            sA[row * 128 + col] = acc[t][j] + bias;
        }
    }
    __syncthreads();

    // ---- softmax over the 16 neighbors (per channel) + weighted sum ----
    float* outp = out + (size_t)r * 128;
#pragma unroll
    for (int cc = 0; cc < 4; ++cc) {
        int c = lane + cc * 32;
        float m = -3.4e38f;
#pragma unroll
        for (int kk = 0; kk < KNN; ++kk) m = fmaxf(m, sA[kk * 128 + c]);
        float w[KNN]; float s = 0.0f;
#pragma unroll
        for (int kk = 0; kk < KNN; ++kk) {
            float e = __expf(sA[kk * 128 + c] - m);
            w[kk] = e; s += e;
        }
        float inv = 1.0f / s;
        float o = 0.0f;
#pragma unroll
        for (int kk = 0; kk < KNN; ++kk) o += w[kk] * inv * sB[kk * 128 + c];
        outp[c] = o;
    }
}

// ---------------------------------------------------------------------------
extern "C" void kernel_launch(void* const* d_in, const int* in_sizes, int n_in,
                              void* d_out, int out_size, void* d_ws, size_t ws_size,
                              hipStream_t stream) {
    const int B = 2, Npts = 8192, D = DIMC;
    const int R = B * Npts;

    const float* xyz   = (const float*)d_in[0];
    const float* feats = (const float*)d_in[1];
    const float* wq    = (const float*)d_in[2];
    const float* bq    = (const float*)d_in[3];
    const float* wk    = (const float*)d_in[4];
    const float* bk    = (const float*)d_in[5];
    const float* wv    = (const float*)d_in[6];
    const float* bv    = (const float*)d_in[7];
    const float* posw1 = (const float*)d_in[8];
    const float* posb1 = (const float*)d_in[9];
    const float* posw2 = (const float*)d_in[10];
    const float* posb2 = (const float*)d_in[11];
    const float* aw1   = (const float*)d_in[12];
    const float* ab1   = (const float*)d_in[13];
    const float* aw2   = (const float*)d_in[14];
    const float* ab2   = (const float*)d_in[15];

    char* w = (char*)d_ws;
    int*      wsIdx  = (int*)w;       w += (size_t)R * KNN * sizeof(int);
    _Float16* featsH = (_Float16*)w;  w += (size_t)R * D * sizeof(_Float16);
    _Float16* wT     = (_Float16*)w;  w += (size_t)6 * D * D * sizeof(_Float16);
    float*    qbuf   = (float*)w;     w += (size_t)R * D * sizeof(float);
    float*    kbuf   = (float*)w;     w += (size_t)R * D * sizeof(float);
    float*    vbuf   = (float*)w;     w += (size_t)R * D * sizeof(float);

    cvt_feats_kernel<<<(R * D + 255) / 256, 256, 0, stream>>>(feats, featsH, R * D);
    cvt_weights_kernel<<<(6 * D * D) / 256, 256, 0, stream>>>(wq, wk, wv, posw2, aw1, aw2, wT);
    knn_kernel<<<R / 256, 256, 0, stream>>>(xyz, wsIdx, Npts);
    qkv_kernel<<<R / 16, 256, 0, stream>>>(featsH, wT, bq, bk, bv, qbuf, kbuf, vbuf);
    pt_attn_kernel<<<R, 32, 0, stream>>>(xyz, wsIdx, qbuf, kbuf, vbuf,
                                         wT + 3 * D * D, wT + 4 * D * D, wT + 5 * D * D,
                                         posw1, posb1, posb2, ab1, ab2,
                                         (float*)d_out, Npts);
}
